// DYAN_Encoder_38800734552141
// MI455X (gfx1250) — compile-verified, matrix operand order checked by reading, not run on hardware
//
#include <hip/hip_runtime.h>
#include <math.h>

typedef __attribute__((ext_vector_type(16))) __bf16 v16bf;
typedef __attribute__((ext_vector_type(8)))  __bf16 v8bf;
typedef __attribute__((ext_vector_type(8)))  float  v8f;

#define T_     10
#define P_     8192
#define K_     645            // 4N+1
#define KP_    656            // 41*16 (DtD padding)
#define KQ_    672            // 21*32 (full padded dimension, rows & K)
#define KT_    21             // k tiles of 32
#define RB_    21             // row blocks of 32
#define CB_    16             // col blocks of 512
#define NITER_ 100
#define NBLK_  (RB_ * CB_)    // 336 fista blocks

// ---------------------------------------------------------------- dictionary
// D stored [16][672] f32, zero padded (rows >= 10, cols >= 645)
__global__ void k_build_dict(const float* __restrict__ rr,
                             const float* __restrict__ th,
                             float* __restrict__ D) {
  int k = blockIdx.x * blockDim.x + threadIdx.x;
  if (k >= KQ_) return;
  float col[T_];
  if (k >= K_) {
    for (int t = 0; t < 16; ++t) D[t * KQ_ + k] = 0.f;
    return;
  }
  if (k == 0) {
    for (int t = 0; t < T_; ++t) col[t] = 1.f;
  } else {
    int grp = (k - 1) / 161;
    int idx = (k - 1) % 161;
    float r = rr[idx], q = th[idx];
    float rp = 1.f;
    for (int t = 0; t < T_; ++t) {
      float ft = (float)t;
      float c = cosf(q * ft), s = sinf(q * ft);
      float sg = (t & 1) ? -1.f : 1.f;
      float v;
      if (grp == 0)      v = rp * c;
      else if (grp == 1) v = sg * rp * c;
      else if (grp == 2) v = rp * s;
      else               v = sg * rp * s;
      col[t] = v;
      rp *= r;
    }
  }
  float n2 = 0.f;
  for (int t = 0; t < T_; ++t) n2 += col[t] * col[t];
  float n = sqrtf(n2);
  if (n == 0.f) n = sqrtf((float)T_);
  for (int t = 0; t < T_; ++t) D[t * KQ_ + k] = col[t] / n;
  for (int t = T_; t < 16; ++t) D[t * KQ_ + k] = 0.f;
}

// ---------------------------------------------------------------- DtD (small)
__global__ void k_dtd(const float* __restrict__ D, float* __restrict__ DtD) {
  int idx = blockIdx.x * blockDim.x + threadIdx.x;
  if (idx >= KP_ * KP_) return;
  int i = idx / KP_, j = idx % KP_;
  float s = 0.f;
  for (int t = 0; t < T_; ++t) s += D[t * KQ_ + i] * D[t * KQ_ + j];
  DtD[idx] = s;
}

// ---------------------------------------------- spectral norm: power iteration
__global__ void k_power(const float* __restrict__ DtD, float* __restrict__ scal) {
  __shared__ float sv[KP_];
  __shared__ float red[1024];
  int i = threadIdx.x;
  float vi = (i < K_) ? 1.f : 0.f;
  float lam = 1.f;
  for (int it = 0; it < 50; ++it) {
    if (i < KP_) sv[i] = vi;
    __syncthreads();
    float s = 0.f;
    if (i < KP_) {
      const float* row = DtD + (size_t)i * KP_;
      for (int j = 0; j < KP_; ++j) s += row[j] * sv[j];
    }
    red[i] = s * s;
    __syncthreads();
    for (int st = 512; st > 0; st >>= 1) {
      if (i < st) red[i] += red[i + st];
      __syncthreads();
    }
    float nrm = sqrtf(red[0]);
    lam = nrm;
    vi = (i < KP_ && nrm > 0.f) ? s / nrm : 0.f;
    __syncthreads();
  }
  if (i == 0) {
    scal[0] = lam;
    scal[1] = 1.f / lam;       // linv
    scal[2] = 0.1f / lam;      // lam = lambda * linv
  }
}

// ------------------------------------------------------- A = I - DtD/L (bf16)
__global__ void k_makeA(const float* __restrict__ DtD,
                        const float* __restrict__ scal,
                        __bf16* __restrict__ A) {
  int idx = blockIdx.x * blockDim.x + threadIdx.x;
  if (idx >= KQ_ * KQ_) return;
  int m = idx / KQ_, kk = idx % KQ_;
  float a = 0.f;
  if (m < K_ && kk < K_)
    a = ((m == kk) ? 1.f : 0.f) - DtD[m * KP_ + kk] * scal[1];
  A[idx] = (__bf16)a;
}

// ---------------------------------------------------------------- DtY * linv
__global__ void k_dty(const float* __restrict__ D, const float* __restrict__ Y,
                      const float* __restrict__ scal, float* __restrict__ DtY) {
  int p = blockIdx.x * blockDim.x + threadIdx.x;
  int m = blockIdx.y;                 // 0..671, D zero-padded -> pad rows = 0
  float s = 0.f;
  for (int t = 0; t < T_; ++t) s += D[t * KQ_ + m] * Y[t * P_ + p];
  DtY[(size_t)m * P_ + p] = s * scal[1];
}

// ------------------------------------------------------- momentum coeffs etc.
__global__ void k_init(float* __restrict__ tts, int* __restrict__ done) {
  if (threadIdx.x == 0 && blockIdx.x == 0) {
    double t = 1.0;
    for (int i = 0; i < NITER_; ++i) {
      double tn = (1.0 + sqrt(1.0 + 4.0 * t * t)) * 0.5;
      tts[i] = (float)((t - 1.0) / tn);
      t = tn;
    }
    done[0] = 0;
  }
}

// ---------------------------------------------------------- main FISTA step
// grid = (16 col-blocks of 512, 21 row-blocks of 32), block = 256 (8 waves)
// wave tile = 32 rows x 64 cols = 2x4 WMMA accumulators
__global__ __launch_bounds__(256) void k_fista(
    const __bf16* __restrict__ A, const __bf16* __restrict__ Yin,
    __bf16* __restrict__ Yout, const float* __restrict__ DtY,
    const float* __restrict__ Xold, float* __restrict__ Xnew,
    const float* __restrict__ tts, const float* __restrict__ scal,
    const int* __restrict__ done, float* __restrict__ diffp, int it) {
  __shared__ __align__(16) __bf16 sA[32 * KQ_];   // 43008 B
  __shared__ float sred[256];
  const int tid  = threadIdx.x;
  const int lane = tid & 31;
  const int w    = tid >> 5;
  const int rb   = blockIdx.y;
  const int cb   = blockIdx.x;
  const int isdone = done[0];
  float ldiff = 0.f;

  if (!isdone) {
    const float tt  = tts[it];
    const float lam = scal[2];

    // stage 32-row A stripe into LDS (coalesced 16B chunks)
    {
      const uint4* gsrc = (const uint4*)(A + (size_t)rb * 32 * KQ_);
      uint4* ldst = (uint4*)sA;
      for (int idx = tid; idx < 32 * KQ_ * 2 / 16; idx += 256)
        ldst[idx] = gsrc[idx];
    }
    __syncthreads();

    v8f acc[2][4];
#pragma unroll
    for (int mt = 0; mt < 2; ++mt)
#pragma unroll
      for (int pt = 0; pt < 4; ++pt) acc[mt][pt] = {};

    // strength-reduced bases
    const __bf16* bp = Yin + ((size_t)(cb * 32 + w * 4)) * 512 + lane * 16;
    const __bf16* ap = sA + (lane & 15) * KQ_ + ((lane >> 4) << 3);

    for (int kt = 0; kt < KT_; ++kt) {
      // B fragments: 4 p-tiles at constant 1KB offsets from one base
      v16bf b0 = *(const v16bf*)(bp + 0 * 512);
      v16bf b1 = *(const v16bf*)(bp + 1 * 512);
      v16bf b2 = *(const v16bf*)(bp + 2 * 512);
      v16bf b3 = *(const v16bf*)(bp + 3 * 512);
      // A fragments for the two 16-row sub-tiles (immediate LDS offsets)
      v8bf a0l = *(const v8bf*)(ap);
      v8bf a0h = *(const v8bf*)(ap + 16);
      v8bf a1l = *(const v8bf*)(ap + 16 * KQ_);
      v8bf a1h = *(const v8bf*)(ap + 16 * KQ_ + 16);
      v16bf a0 = __builtin_shufflevector(a0l, a0h, 0, 1, 2, 3, 4, 5, 6, 7, 8,
                                         9, 10, 11, 12, 13, 14, 15);
      v16bf a1 = __builtin_shufflevector(a1l, a1h, 0, 1, 2, 3, 4, 5, 6, 7, 8,
                                         9, 10, 11, 12, 13, 14, 15);
      acc[0][0] = __builtin_amdgcn_wmma_f32_16x16x32_bf16(false, a0, false, b0,
                                                          (short)0, acc[0][0], false, false);
      acc[0][1] = __builtin_amdgcn_wmma_f32_16x16x32_bf16(false, a0, false, b1,
                                                          (short)0, acc[0][1], false, false);
      acc[0][2] = __builtin_amdgcn_wmma_f32_16x16x32_bf16(false, a0, false, b2,
                                                          (short)0, acc[0][2], false, false);
      acc[0][3] = __builtin_amdgcn_wmma_f32_16x16x32_bf16(false, a0, false, b3,
                                                          (short)0, acc[0][3], false, false);
      acc[1][0] = __builtin_amdgcn_wmma_f32_16x16x32_bf16(false, a1, false, b0,
                                                          (short)0, acc[1][0], false, false);
      acc[1][1] = __builtin_amdgcn_wmma_f32_16x16x32_bf16(false, a1, false, b1,
                                                          (short)0, acc[1][1], false, false);
      acc[1][2] = __builtin_amdgcn_wmma_f32_16x16x32_bf16(false, a1, false, b2,
                                                          (short)0, acc[1][2], false, false);
      acc[1][3] = __builtin_amdgcn_wmma_f32_16x16x32_bf16(false, a1, false, b3,
                                                          (short)0, acc[1][3], false, false);
      bp += 512 * 512;   // next 32-K slab of y
      ap += 32;          // next 32 K-columns of A
    }

    // epilogue: x_new = shrink(Ay + DtY); y_new written as full B fragments
    const int colbase = cb * 512 + w * 64;
    const int rhalf   = (lane >> 4) << 3;
    const int coln    = colbase + (lane & 15);
#pragma unroll
    for (int pt = 0; pt < 4; ++pt) {
      float yv[16];
#pragma unroll
      for (int mt = 0; mt < 2; ++mt)
#pragma unroll
        for (int r = 0; r < 8; ++r) {
          const int row = rb * 32 + mt * 16 + r + rhalf;
          const size_t idx = (size_t)row * P_ + coln + pt * 16;
          float z  = acc[mt][pt][r] + DtY[idx];
          float xn = (z > lam) ? (z - lam) : ((z < -lam) ? (z + lam) : 0.f);
          float xo = Xold[idx];
          Xnew[idx] = xn;
          float d = xn - xo;
          ldiff += d * d;
          yv[mt * 8 + r] = xn * (1.f + tt) - xo * tt;
        }
      // wave's 32 rows == exactly K-tile rb of next iteration's B operand:
      // acc layout maps element-for-element onto the B fragment layout
      v16bf yb;
#pragma unroll
      for (int e = 0; e < 16; ++e) yb[e] = (__bf16)yv[e];
      const int ptg = cb * 32 + w * 4 + pt;
      *(v16bf*)(Yout + ((size_t)(rb * 512 + ptg)) * 512 + lane * 16) = yb;
    }
  } else {
    // frozen: carry x forward for ping-pong consistency
    const int colbase = cb * 512 + w * 64;
    const int rhalf   = (lane >> 4) << 3;
    const int coln    = colbase + (lane & 15);
    for (int pt = 0; pt < 4; ++pt)
      for (int mt = 0; mt < 2; ++mt)
        for (int r = 0; r < 8; ++r) {
          const int row = rb * 32 + mt * 16 + r + rhalf;
          const size_t idx = (size_t)row * P_ + coln + pt * 16;
          Xnew[idx] = Xold[idx];
        }
  }

  // deterministic per-block diff partial
  sred[tid] = ldiff;
  __syncthreads();
  for (int st = 128; st > 0; st >>= 1) {
    if (tid < st) sred[tid] += sred[tid + st];
    __syncthreads();
  }
  if (tid == 0) diffp[blockIdx.y * gridDim.x + blockIdx.x] = sred[0];
}

// --------------------------------------------------- sticky early-stop check
__global__ void k_done(const float* __restrict__ diffp, int* __restrict__ done) {
  if (threadIdx.x == 0 && blockIdx.x == 0) {
    float s = 0.f;
    for (int i = 0; i < NBLK_; ++i) s += diffp[i];
    if (!done[0] && (sqrtf(s) * (1.f / (float)K_) < 1e-4f)) done[0] = 1;
  }
}

// ============================================================================
extern "C" void kernel_launch(void* const* d_in, const int* in_sizes, int n_in,
                              void* d_out, int out_size, void* d_ws,
                              size_t ws_size, hipStream_t stream) {
  (void)in_sizes; (void)n_in; (void)ws_size; (void)out_size;
  const float* Y   = (const float*)d_in[0];   // [1,10,8192]
  const float* Drr = (const float*)d_in[1];   // [161]
  const float* Dth = (const float*)d_in[2];   // [161]

  char* ws = (char*)d_ws;
  size_t off = 0;
  auto alloc = [&](size_t bytes) {
    size_t o = off;
    off = (off + bytes + 255) & ~(size_t)255;
    return o;
  };
  float*  D    = (float*)(ws + alloc((size_t)16 * KQ_ * 4));
  float*  DtD  = (float*)(ws + alloc((size_t)KP_ * KP_ * 4));
  __bf16* Abf  = (__bf16*)(ws + alloc((size_t)KQ_ * KQ_ * 2));
  float*  DtY  = (float*)(ws + alloc((size_t)KQ_ * P_ * 4));
  float*  X0   = (float*)(ws + alloc((size_t)KQ_ * P_ * 4));
  float*  X1   = (float*)(ws + alloc((size_t)KQ_ * P_ * 4));
  const size_t ybytes = (size_t)KT_ * 512 * 512 * 2;  // 21 ktiles * 512 ptiles * 1KB
  __bf16* Yf0  = (__bf16*)(ws + alloc(ybytes));
  __bf16* Yf1  = (__bf16*)(ws + alloc(ybytes));
  float*  tts  = (float*)(ws + alloc(128 * 4));
  float*  diffp= (float*)(ws + alloc((size_t)NBLK_ * 4));
  float*  scal = (float*)(ws + alloc(16 * 4));
  int*    done = (int*)(ws + alloc(16 * 4));

  // x0 = 0, y0 = 0 (both y buffers: pad rows must stay zero)
  hipMemsetAsync(X0,  0, (size_t)KQ_ * P_ * 4, stream);
  hipMemsetAsync(Yf0, 0, ybytes, stream);
  hipMemsetAsync(Yf1, 0, ybytes, stream);

  k_build_dict<<<(KQ_ + 255) / 256, 256, 0, stream>>>(Drr, Dth, D);
  k_dtd<<<(KP_ * KP_ + 255) / 256, 256, 0, stream>>>(D, DtD);
  k_power<<<1, 1024, 0, stream>>>(DtD, scal);
  k_makeA<<<(KQ_ * KQ_ + 255) / 256, 256, 0, stream>>>(DtD, scal, Abf);
  k_dty<<<dim3(P_ / 256, KQ_), 256, 0, stream>>>(D, Y, scal, DtY);
  k_init<<<1, 32, 0, stream>>>(tts, done);

  for (int it = 0; it < NITER_; ++it) {
    const float*  Xo = (it & 1) ? X1 : X0;
    float*        Xn = (it & 1) ? X0 : X1;
    const __bf16* Yi = (it & 1) ? Yf1 : Yf0;
    __bf16*       Yo = (it & 1) ? Yf0 : Yf1;
    k_fista<<<dim3(CB_, RB_), 256, 0, stream>>>(Abf, Yi, Yo, DtY, Xo, Xn, tts,
                                                scal, done, diffp, it);
    k_done<<<1, 32, 0, stream>>>(diffp, done);
  }

  // 100 iters -> final x lives in X0; first 645 rows are the output (contiguous)
  hipMemcpyAsync(d_out, X0, (size_t)K_ * P_ * 4, hipMemcpyDeviceToDevice,
                 stream);
}
